// KANLayer_75814762709066
// MI455X (gfx1250) — matrix operand order repaired and checked by reading, not compiled, not run on hardware
//
#include <hip/hip_runtime.h>

typedef __attribute__((ext_vector_type(8)))  _Float16 v8h;
typedef __attribute__((ext_vector_type(16))) _Float16 v16h;
typedef __attribute__((ext_vector_type(8)))  float    v8f;

#define IN_F   512
#define OUT_F  512
#define NKNOT  32
#define BATCH  2048

// ---------------------------------------------------------------------------
// gfx1250 async copy: global -> LDS, 16 bytes, tracked by ASYNCcnt.
// LDS destination address = low 32 bits of the generic shared pointer
// ---------------------------------------------------------------------------
__device__ __forceinline__ void async_b128(void* lds_dst, const void* gsrc) {
    asm volatile("global_load_async_to_lds_b128 %0, %1, off"
                 :
                 : "v"((unsigned int)(unsigned long long)lds_dst),
                   "v"((unsigned long long)gsrc)
                 : "memory");
}
__device__ __forceinline__ void wait_async0() {
    asm volatile("s_wait_asynccnt 0x0" ::: "memory");
}

// ---------------------------------------------------------------------------
// Kernel 1: coeffs (i, o, k) f32  ->  W[(i*32+k)*512 + o] f16
// ---------------------------------------------------------------------------
__global__ __launch_bounds__(256) void convert_w(const float* __restrict__ C,
                                                 _Float16* __restrict__ W) {
    const int d  = (blockIdx.x * 256 + threadIdx.x) * 8;   // < 8388608
    const int o  = d & 511;
    const int ik = d >> 9;
    const int i  = ik >> 5;
    const int k  = ik & 31;
    const float* src = C + i * (OUT_F * NKNOT) + o * NKNOT + k;
    v8h w;
#pragma unroll
    for (int t = 0; t < 8; ++t) w[t] = (_Float16)src[t * NKNOT];
    *(v8h*)(W + d) = w;
}

// ---------------------------------------------------------------------------
// Kernel 2: fused basis generation + WMMA GEMM, double-buffered LDS,
// async global->LDS staging of the W tile.
//   out[b,o] = sum_{i,k} basis[b,i,k] * W[(i,k),o]
// Block: 256 threads (8 waves), output tile 64(M) x 64(N), K-chunk 64.
// ---------------------------------------------------------------------------
__global__ __launch_bounds__(256) void kan_gemm(const float* __restrict__ X,
                                                const _Float16* __restrict__ W,
                                                const float* __restrict__ Kn,
                                                float* __restrict__ Out) {
    const int tid  = threadIdx.x;
    const int lane = tid & 31;
    const int wave = tid >> 5;
    const int wm   = wave & 3;   // M fragment: rows wm*16..+15
    const int wn   = wave >> 2;  // N half:     cols wn*32..+31
    const int m0   = blockIdx.x * 64;
    const int o0   = blockIdx.y * 64;

    // A slab: k permuted into WMMA fragment order per 32-k row:
    // groups [k0-7, k16-23, k8-15, k24-31] -> lane reads contiguous v16h.
    __shared__ __align__(32) _Float16 Ash[2 * 4096];  // [buf][ii*2048 + r*32 + perm]
    __shared__ __align__(32) _Float16 Bsh[2 * 4096];  // [buf][ii*2048 + k*64 + o]

    const float kn0   = Kn[0];
    const float h     = Kn[1] - kn0;
    const float inv_h = 1.0f / h;

    v8f acc0 = {};
    v8f acc1 = {};

    const int mloc  = lane & 15;
    const int khalf = lane >> 4;

    // Stage K-chunk starting at feature i0 into buffer `buf`.
    auto stage = [&](int i0, int buf) {
        // ---- B tile: async global -> LDS, 16 f16 per thread ----
        {
            const int ii  = tid >> 7;
            const int lcl = tid & 127;
            const int k   = lcl >> 2;
            const int ob  = (lcl & 3) << 4;  // 0,16,32,48
            const _Float16* g = W + ((i0 + ii) * NKNOT + k) * OUT_F + o0 + ob;
            _Float16* d = &Bsh[buf * 4096 + ii * 2048 + k * 64 + ob];
            async_b128(d, g);
            async_b128(d + 8, g + 8);
        }
        // ---- A tile: degree-3 Cox-de Boor basis (4 nonzeros per row) ----
        if (tid < 128) {
            const int ii = tid >> 6;
            const int r  = tid & 63;
            const float x = X[(m0 + r) * IN_F + (i0 + ii)];
            _Float16* row = &Ash[buf * 4096 + ii * 2048 + r * 32];
            v8h z = {};
            ((v8h*)row)[0] = z; ((v8h*)row)[1] = z;
            ((v8h*)row)[2] = z; ((v8h*)row)[3] = z;

            int j = (int)floorf((x - kn0) * inv_h);
            j = j < 0 ? 0 : (j > 31 ? 31 : j);
            const float tj = kn0 + (float)j * h;
            const float xm_tj   = x - tj;           // x - t_j
            const float tj1_x   = (tj + h) - x;     // t_{j+1} - x
            const float xm_tjm1 = xm_tj + h;        // x - t_{j-1}
            const float tj2_x   = tj1_x + h;        // t_{j+2} - x
            const float xm_tjm2 = xm_tj + 2.0f * h; // x - t_{j-2}
            const float tj3_x   = tj1_x + 2.0f * h; // t_{j+3} - x
            const float d1 = inv_h, d2 = 0.5f * inv_h, d3 = (1.0f / 3.0f) * inv_h;
            const float b1_0 = tj1_x * d1;                            // N1_{j-1}
            const float b1_1 = xm_tj * d1;                            // N1_{j}
            const float b2_0 = tj1_x * d2 * b1_0;                     // N2_{j-2}
            const float b2_1 = (xm_tjm1 * b1_0 + tj2_x * b1_1) * d2;  // N2_{j-1}
            const float b2_2 = xm_tj * d2 * b1_1;                     // N2_{j}
            float bb[4];
            bb[0] = tj1_x * d3 * b2_0;                                // N3_{j-3}
            bb[1] = (xm_tjm2 * b2_0 + tj2_x * b2_1) * d3;             // N3_{j-2}
            bb[2] = (xm_tjm1 * b2_1 + tj3_x * b2_2) * d3;             // N3_{j-1}
            bb[3] = xm_tj * d3 * b2_2;                                // N3_{j}
#pragma unroll
            for (int s = 0; s < 4; ++s) {
                const int idx = j - 3 + s;
                if (idx >= 0 && idx < 32) {
                    const int g   = idx >> 3;
                    const int pos = ((((g << 1) | (g >> 1)) & 3) << 3) | (idx & 7);
                    row[pos] = (_Float16)bb[s];
                }
            }
        }
    };

    // Prologue: stage chunk 0 into buffer 0.
    stage(0, 0);
    wait_async0();
    __syncthreads();

    for (int it = 0; it < IN_F / 2; ++it) {
        const int cur = it & 1;
        // Stage next chunk into the other buffer while we do the math.
        if (it + 1 < IN_F / 2) stage((it + 1) << 1, cur ^ 1);

        // ---- WMMA on current buffer: 2 K-steps x 2 N-frags per wave ----
#pragma unroll
        for (int ii = 0; ii < 2; ++ii) {
            const _Float16* arow =
                &Ash[cur * 4096 + ii * 2048 + (wm * 16 + mloc) * 32 + khalf * 16];
            const v16h a = *(const v16h*)arow;
            const _Float16* brow = &Bsh[cur * 4096 + ii * 2048 + khalf * 8 * 64 + wn * 32 + mloc];
            v16h b0, b1;
#pragma unroll
            for (int s = 0; s < 16; ++s) {
                const int kk = s < 8 ? s : s + 8;
                b0[s] = brow[kk * 64];
                b1[s] = brow[kk * 64 + 16];
            }
            acc0 = __builtin_amdgcn_wmma_f32_16x16x32_f16(
                false, a, false, b0, (short)0, acc0, false, false);
            acc1 = __builtin_amdgcn_wmma_f32_16x16x32_f16(
                false, a, false, b1, (short)0, acc1, false, false);
        }

        wait_async0();
        __syncthreads();
    }

    // ---- store C (16x16 f32 D layout: VGPR r -> row 8*(lane/16)+r, lane%16 -> col) ----
    const int nbase = o0 + wn * 32 + (lane & 15);
    const int mbase = m0 + wm * 16 + ((lane >> 4) << 3);
#pragma unroll
    for (int r = 0; r < 8; ++r) {
        Out[(mbase + r) * OUT_F + nbase]      = acc0[r];
        Out[(mbase + r) * OUT_F + nbase + 16] = acc1[r];
    }
}

extern "C" void kernel_launch(void* const* d_in, const int* in_sizes, int n_in,
                              void* d_out, int out_size, void* d_ws, size_t ws_size,
                              hipStream_t stream) {
    const float* x      = (const float*)d_in[0];
    const float* coeffs = (const float*)d_in[1];
    const float* knots  = (const float*)d_in[2];
    float* out = (float*)d_out;
    _Float16* W = (_Float16*)d_ws;  // 512*512*32 f16 = 16 MB scratch

    // 8388608 elements / 8 per thread / 256 per block = 4096 blocks
    convert_w<<<4096, 256, 0, stream>>>(coeffs, W);

    dim3 grid(BATCH / 64, OUT_F / 64);  // (32, 8)
    kan_gemm<<<grid, 256, 0, stream>>>(x, W, knots, out);
}
